// ContinuousGRULayer_10660108829183
// MI455X (gfx1250) — matrix operand order repaired and checked by reading, not compiled, not run on hardware
//
#include <hip/hip_runtime.h>
#include <hip/hip_bf16.h>

// ---------------------------------------------------------------------------
// Continuous GRU (GRU-ODE, RK4) for MI455X / gfx1250.
// 8 persistent workgroups x 16 batch rows; recurrence is per-batch-row
// independent -> no cross-WGP sync. bf16 weights (1.97MB) stay hot in L2.
// v_wmma_f32_16x16x32_bf16 with f32 accumulate; h / RK4 state f32 in LDS.
// Phase 1: each wave computes the z-tile AND r-tile for one column tile,
// sharing one A fragment per k-step (2 WMMAs per A load).
// The weight *offset* (not pointer) is laundered through empty asm each eval:
// loads stay in-loop as global_load_b128 (L2 hot, addrspace preserved),
// and LICM cannot hoist 320 VGPRs of weights into a scratch spill.
// ---------------------------------------------------------------------------

#define Bsz 128
#define Tsz 512
#define Fsz 128
#define Hsz 512
#define Ksz 640            // F + H
#define KBLK 20            // 640 / 32
#define ROWS 16            // batch rows per workgroup
#define NWAVES 32          // 1024 threads
#define APAD 648           // bf16 A-buffer row stride (1296B = 324 dw, rotates banks)
#define CP 516             // f32 state row stride (pad banks)
#define WN (Hsz * Ksz)     // elements per weight matrix

typedef __bf16 v16bf __attribute__((ext_vector_type(16)));
typedef float  v8f   __attribute__((ext_vector_type(8)));

// A fragment (16x32 bf16) from LDS: lane<16 -> K {0..7,16..23}; lane>=16 -> {8..15,24..31}
__device__ __forceinline__ void load_afrag(v16bf* a, const __bf16* abase, int kb) {
  ((uint4*)a)[0] = *(const uint4*)(abase + kb * 32);
  ((uint4*)a)[1] = *(const uint4*)(abase + kb * 32 + 16);
}
// B fragment (32x16 bf16) from global: lane&15 = N column, lane<16 -> K {0..7,16..23}; lane>=16 -> {8..15,24..31}
__device__ __forceinline__ void load_bfrag(v16bf* b, const __bf16* wbase, int kb) {
  ((uint4*)b)[0] = *(const uint4*)(wbase + kb * 32);
  ((uint4*)b)[1] = *(const uint4*)(wbase + kb * 32 + 16);
}

__device__ __forceinline__ v8f wmma_bf16(v16bf a, v16bf b, v8f c) {
  return __builtin_amdgcn_wmma_f32_16x16x32_bf16(false, a, false, b,
                                                 (short)0, c, false, false);
}

__device__ __forceinline__ float sigmoidf_(float v) {
  return 1.0f / (1.0f + __expf(-v));
}

__global__ __launch_bounds__(1024, 1)
void gru_ode_persistent(const float* __restrict__ x,      // [B,T,F]
                        const float* __restrict__ tdif,   // [B,T]
                        const float* __restrict__ bz,
                        const float* __restrict__ br,
                        const float* __restrict__ bg,
                        const __bf16* __restrict__ Wz,    // [H,K] bf16
                        const __bf16* __restrict__ Wr,
                        const __bf16* __restrict__ Wg,
                        float* __restrict__ out) {        // [B,T,H]
  __shared__ float s_h [ROWS * CP];   // recurrent state (f32)
  __shared__ float s_ka[ROWS * CP];   // RK4 k accumulator
  __shared__ float s_z [ROWS * CP];   // z gate for current eval
  __shared__ __align__(16) __bf16 s_A1[ROWS * APAD];  // [x | h_eval]   (bf16)
  __shared__ __align__(16) __bf16 s_A2[ROWS * APAD];  // [x | r*h_eval] (bf16)
  __shared__ float s_bz[Hsz], s_br[Hsz], s_bg[Hsz];
  __shared__ float s_dt[ROWS];

  const int tid   = threadIdx.x;
  const int wid   = tid >> 5;          // 0..31 == column tile index
  const int lane  = tid & 31;
  const int m     = lane & 15;
  const int hi    = lane >> 4;
  const int bbase = blockIdx.x * ROWS;

  // per-wave fragment base pointers / offsets (fixed for whole kernel)
  const __bf16* abase1 = s_A1 + m * APAD + hi * 8;      // A frag base, buffer 1
  const __bf16* abase2 = s_A2 + m * APAD + hi * 8;      // A frag base, buffer 2
  const unsigned woff  = (unsigned)((wid * 16 + m) * Ksz + hi * 8);
  const int col   = wid * 16 + m;      // output column this lane owns
  const int rbase = hi * 8;            // output row base this lane owns

  // ---- one-time init: biases, h = 0, A1 h-part = 0
  for (int i = tid; i < Hsz; i += 1024) {
    s_bz[i] = bz[i]; s_br[i] = br[i]; s_bg[i] = bg[i];
  }
  for (int i = tid; i < ROWS * Hsz; i += 1024) {
    int row = i >> 9, c = i & (Hsz - 1);
    s_h[row * CP + c] = 0.0f;
    s_A1[row * APAD + Fsz + c] = (__bf16)0.0f;
  }
  __syncthreads();

  for (int t = 0; t < Tsz; ++t) {
    // ---- per-timestep prologue: x_t -> bf16 into both A buffers, dt
    for (int i = tid; i < ROWS * Fsz; i += 1024) {
      int row = i >> 7, c = i & (Fsz - 1);
      float v = x[((size_t)(bbase + row) * Tsz + t) * Fsz + c];
      __bf16 bv = (__bf16)v;
      s_A1[row * APAD + c] = bv;
      s_A2[row * APAD + c] = bv;
    }
    if (tid < ROWS) {
      float v = tdif[(size_t)(bbase + tid) * Tsz + t];
      s_dt[tid] = fminf(v, 1.0f) * 0.5f;   // MAX_DT / ODE_STEPS
    }
    __syncthreads();

    for (int sub = 0; sub < 2; ++sub) {       // ODE_STEPS
      for (int e = 0; e < 4; ++e) {           // RK4 stages
        // ===== phase 1: z-tile and r-tile for column tile `wid` ===========
        // 2 WMMAs per shared A fragment; software-pipelined operand loads.
        {
          // launder the element offset (32-bit): loads stay in-loop and keep
          // global addrspace (global_load_b128 from L2-hot weights, no spill)
          unsigned o = woff;
          asm volatile("" : "+v"(o));
          const __bf16* wz_p = Wz + o;
          const __bf16* wr_p = Wr + o;

          v8f accZ = {}, accR = {};
          v16bf a0, z0, r0, a1, z1, r1;
          load_afrag(&a0, abase1, 0);
          load_bfrag(&z0, wz_p, 0);
          load_bfrag(&r0, wr_p, 0);
#pragma unroll
          for (int kb = 0; kb < KBLK; ++kb) {
            if (kb + 1 < KBLK) {
              load_afrag(&a1, abase1, kb + 1);
              load_bfrag(&z1, wz_p, kb + 1);
              load_bfrag(&r1, wr_p, kb + 1);
            }
            accZ = wmma_bf16(a0, z0, accZ);
            accR = wmma_bf16(a0, r0, accR);
            a0 = a1; z0 = z1; r0 = r1;
          }
#pragma unroll
          for (int r8 = 0; r8 < 8; ++r8) {
            const int row = rbase + r8;
            float zv = sigmoidf_(accZ[r8] + s_bz[col]);
            float rv = sigmoidf_(accR[r8] + s_br[col]);
            float he = (float)s_A1[row * APAD + Fsz + col];
            s_z[row * CP + col] = zv;
            s_A2[row * APAD + Fsz + col] = (__bf16)(rv * he);
          }
        }
        __syncthreads();

        // ===== phase 2: g-tile for column tile `wid` + RK4 epilogue =======
        {
          unsigned o = woff;
          asm volatile("" : "+v"(o));
          const __bf16* wg_p = Wg + o;

          v8f acc = {};
          v16bf a0, g0, a1, g1;
          load_afrag(&a0, abase2, 0);
          load_bfrag(&g0, wg_p, 0);
#pragma unroll
          for (int kb = 0; kb < KBLK; ++kb) {
            if (kb + 1 < KBLK) {
              load_afrag(&a1, abase2, kb + 1);
              load_bfrag(&g1, wg_p, kb + 1);
            }
            acc = wmma_bf16(a0, g0, acc);
            a0 = a1; g0 = g1;
          }
#pragma unroll
          for (int r8 = 0; r8 < 8; ++r8) {
            const int row = rbase + r8;
            float g  = tanhf(acc[r8] + s_bg[col]);
            float he = (float)s_A1[row * APAD + Fsz + col]; // current h_eval
            float zv = s_z[row * CP + col];
            float k  = (1.0f - zv) * (g - he);              // dh/dt
            float dt = s_dt[row];
            float hb = s_h[row * CP + col];
            float hen;
            if (e == 0)      { s_ka[row * CP + col] = k;            hen = hb + 0.5f * dt * k; }
            else if (e == 1) { s_ka[row * CP + col] += 2.0f * k;    hen = hb + 0.5f * dt * k; }
            else if (e == 2) { s_ka[row * CP + col] += 2.0f * k;    hen = hb + dt * k;        }
            else {
              float ks = s_ka[row * CP + col] + k;
              float hn = hb + (dt * (1.0f / 6.0f)) * ks;
              s_h[row * CP + col] = hn;
              hen = hn;
              if (sub == 1)
                out[((size_t)(bbase + row) * Tsz + t) * Hsz + col] = hn;
            }
            s_A1[row * APAD + Fsz + col] = (__bf16)hen;     // next h_eval operand
          }
        }
        __syncthreads();
      }
    }
  }
}

// ---- one-time (per launch) f32 -> bf16 weight conversion into workspace ----
__global__ void convert_weights(const float* __restrict__ Wz,
                                const float* __restrict__ Wr,
                                const float* __restrict__ Wg,
                                __bf16* __restrict__ dst) {
  int i = blockIdx.x * blockDim.x + threadIdx.x;
  if (i < WN) {
    dst[i]          = (__bf16)Wz[i];
    dst[WN + i]     = (__bf16)Wr[i];
    dst[2 * WN + i] = (__bf16)Wg[i];
  }
}

extern "C" void kernel_launch(void* const* d_in, const int* in_sizes, int n_in,
                              void* d_out, int out_size, void* d_ws, size_t ws_size,
                              hipStream_t stream) {
  const float* x  = (const float*)d_in[0];
  const float* td = (const float*)d_in[1];
  const float* Wz = (const float*)d_in[2];
  const float* bz = (const float*)d_in[3];
  const float* Wr = (const float*)d_in[4];
  const float* br = (const float*)d_in[5];
  const float* Wg = (const float*)d_in[6];
  const float* bg = (const float*)d_in[7];

  __bf16* wbf = (__bf16*)d_ws;   // 3 * 512*640 bf16 = 1.97 MB (L2-resident)

  convert_weights<<<(WN + 255) / 256, 256, 0, stream>>>(Wz, Wr, Wg, wbf);

  gru_ode_persistent<<<Bsz / ROWS, NWAVES * 32, 0, stream>>>(
      x, td, bz, br, bg, wbf, wbf + WN, wbf + 2 * WN, (float*)d_out);
}
